// LSTM_21225728377347
// MI455X (gfx1250) — matrix operand order repaired and checked by reading, not compiled, not run on hardware
//
#include <hip/hip_runtime.h>
#include <stdint.h>

#define B_   64
#define T_   512
#define I_   256
#define H_   1024
#define G4H  (4 * H_)
#define NWG  64
#define BLK  128

// LDS row strides (bf16 elements), padded so 16 lanes hit distinct banks:
// WHH/WFC: 1024+24 -> 524 dwords, 524%64=12 (12n mod 64 distinct for n<16)
// WIH:      256+8  -> 132 dwords, 132%64=4  (4n  mod 64 distinct for n<16)
#define WHH_STRIDE 1048
#define WIH_STRIDE 264
#define WFC_STRIDE 1048
#define SMEM_ELEMS (64 * WHH_STRIDE + 64 * WIH_STRIDE + 16 * WFC_STRIDE)
#define SMEM_BYTES (SMEM_ELEMS * 2 + 256 * 4 /* red */)

typedef __attribute__((ext_vector_type(16))) __bf16       v16bf;
typedef __attribute__((ext_vector_type(8)))  float        v8f;
typedef __attribute__((ext_vector_type(4)))  unsigned int u32x4;
typedef __attribute__((ext_vector_type(8)))  unsigned int u32x8;

__device__ __forceinline__ unsigned short f2bf(float f) {
  unsigned u = __builtin_bit_cast(unsigned, f);
  return (unsigned short)((u + 0x7fffu + ((u >> 16) & 1u)) >> 16);
}

__device__ __forceinline__ v16bf pack16(u32x4 lo, u32x4 hi) {
  u32x8 c;
  c[0] = lo[0]; c[1] = lo[1]; c[2] = lo[2]; c[3] = lo[3];
  c[4] = hi[0]; c[5] = hi[1]; c[6] = hi[2]; c[7] = hi[3];
  return __builtin_bit_cast(v16bf, c);
}

// A-fragment 16x32 bf16 (ISA 7.12.2): two 16B row-contiguous chunks at K and K+16.
__device__ __forceinline__ v16bf loadA(const unsigned short* p) {
  u32x4 lo = *(const u32x4*)(p);
  u32x4 hi = *(const u32x4*)(p + 16);
  return pack16(lo, hi);
}

// B-fragment 32x16 bf16: lane holds column N=lane&15, two 8-K chunks at K and K+16.
__device__ __forceinline__ v16bf loadB16(const unsigned short* p) {
  u32x4 lo = *(const u32x4*)(p);
  u32x4 hi = *(const u32x4*)(p + 16);
  return pack16(lo, hi);
}

__device__ __forceinline__ float sigm(float x) {
  return 1.0f / (1.0f + __expf(-x));
}

// Async global->LDS copy, 16B per lane; tracked by ASYNCcnt (CDNA5 ISA).
__device__ __forceinline__ void async_copy_b128(unsigned lds_byte_off, const void* g) {
  asm volatile("global_load_async_to_lds_b128 %0, %1, off"
               :
               : "v"(lds_byte_off), "v"((unsigned long long)(size_t)g)
               : "memory");
}
__device__ __forceinline__ void wait_async0() {
  asm volatile("s_wait_asynccnt 0" ::: "memory");
}

__device__ __forceinline__ void grid_barrier(unsigned* cnt, unsigned* gen) {
  __syncthreads();
  if (threadIdx.x == 0) {
    __threadfence();
    unsigned g = __hip_atomic_load(gen, __ATOMIC_RELAXED, __HIP_MEMORY_SCOPE_AGENT);
    unsigned arrived = __hip_atomic_fetch_add(cnt, 1u, __ATOMIC_ACQ_REL, __HIP_MEMORY_SCOPE_AGENT);
    if (arrived == NWG - 1u) {
      __hip_atomic_store(cnt, 0u, __ATOMIC_RELAXED, __HIP_MEMORY_SCOPE_AGENT);
      __hip_atomic_fetch_add(gen, 1u, __ATOMIC_RELEASE, __HIP_MEMORY_SCOPE_AGENT);
    } else {
      while (__hip_atomic_load(gen, __ATOMIC_ACQUIRE, __HIP_MEMORY_SCOPE_AGENT) == g) {
        __builtin_amdgcn_s_sleep(1);
      }
    }
  }
  __syncthreads();
}

__global__ void k_cvt_bf16(const float* __restrict__ src, unsigned short* __restrict__ dst, int n) {
  int i = blockIdx.x * blockDim.x + threadIdx.x;
  if (i < n) dst[i] = f2bf(src[i]);
}

__global__ void k_prep(const float* __restrict__ b_ih, const float* __restrict__ b_hh,
                       float* __restrict__ bias,
                       const float* __restrict__ c0, float* __restrict__ c_state,
                       const float* __restrict__ h0, unsigned short* __restrict__ hA,
                       unsigned* __restrict__ bar) {
  int i = blockIdx.x * blockDim.x + threadIdx.x;
  if (i < G4H) bias[i] = b_ih[i] + b_hh[i];
  if (i < B_ * H_) { c_state[i] = c0[i]; hA[i] = f2bf(h0[i]); }
  if (i < 2) bar[i] = 0u;
}

__global__ void __launch_bounds__(BLK) lstm_fused(
    const unsigned short* __restrict__ xb,    // [B, T, I] bf16
    const unsigned short* __restrict__ Wih,   // [4H, I]   bf16 (== B^T)
    const unsigned short* __restrict__ Whh,   // [4H, H]   bf16 (== B^T)
    const unsigned short* __restrict__ Wfc,   // [I,  H]   bf16 (== B^T)
    const float* __restrict__ bias,           // [4H] = b_ih + b_hh
    const float* __restrict__ bfc,            // [I]
    float* __restrict__ c_state,              // [B, H] f32
    unsigned short* __restrict__ hA,          // [B, H] bf16 ping
    unsigned short* __restrict__ hB,          // [B, H] bf16 pong
    float* __restrict__ out,                  // [T, B, I] f32
    unsigned* __restrict__ bar)
{
  const int wg    = blockIdx.x;          // 0..63 -> 16-column slice of H
  const int wave  = threadIdx.x >> 5;    // 0..3  -> 16-row batch tile
  const int lane  = threadIdx.x & 31;
  const int lhalf = lane >> 4;
  const int l15   = lane & 15;
  const int nbase = wg * 16;
  const int rt    = wave;
  const int arow  = rt * 16 + l15;       // batch row this lane loads for A
  const int klA   = lhalf * 8;           // A fragment K skew
  const int klB   = lhalf * 8;           // B fragment K skew
  const int frt   = wg >> 4, fct = wg & 15;  // FC output tile of this WG

  extern __shared__ unsigned short smem[];
  unsigned short* whh_sh = smem;                               // 64 * WHH_STRIDE
  unsigned short* wih_sh = whh_sh + 64 * WHH_STRIDE;           // 64 * WIH_STRIDE
  unsigned short* wfc_sh = wih_sh + 64 * WIH_STRIDE;           // 16 * WFC_STRIDE
  float*          red    = (float*)(wfc_sh + 16 * WFC_STRIDE); // 256 f32

  // ---- One-time async stage of this WG's weight slices into LDS ----
  {
    const unsigned whh_off = (unsigned)(size_t)whh_sh;
    const unsigned wih_off = (unsigned)(size_t)wih_sh;
    const unsigned wfc_off = (unsigned)(size_t)wfc_sh;
    // W_hh slice: 64 rows (4 gates x 16 cols) x 1024 bf16 = 128 x 16B chunks/row
    for (int cidx = threadIdx.x; cidx < 64 * 128; cidx += BLK) {
      const int row = cidx >> 7, chunk = cidx & 127;
      const int g = row >> 4, n = row & 15;
      const unsigned short* src = Whh + (size_t)(g * H_ + nbase + n) * H_ + chunk * 8;
      async_copy_b128(whh_off + (unsigned)(row * WHH_STRIDE + chunk * 8) * 2u, src);
    }
    // W_ih slice: 64 rows x 256 bf16 = 32 x 16B chunks/row
    for (int cidx = threadIdx.x; cidx < 64 * 32; cidx += BLK) {
      const int row = cidx >> 5, chunk = cidx & 31;
      const int g = row >> 4, n = row & 15;
      const unsigned short* src = Wih + (size_t)(g * H_ + nbase + n) * I_ + chunk * 8;
      async_copy_b128(wih_off + (unsigned)(row * WIH_STRIDE + chunk * 8) * 2u, src);
    }
    // W_fc slice for FC tile fct: 16 rows x 1024 bf16
    for (int cidx = threadIdx.x; cidx < 16 * 128; cidx += BLK) {
      const int row = cidx >> 7, chunk = cidx & 127;
      const unsigned short* src = Wfc + (size_t)(fct * 16 + row) * H_ + chunk * 8;
      async_copy_b128(wfc_off + (unsigned)(row * WFC_STRIDE + chunk * 8) * 2u, src);
    }
    wait_async0();
  }
  __syncthreads();

  unsigned* cnt = bar;
  unsigned* gen = bar + 1;

  const int ncol = nbase + l15;          // h-space column owned by this lane
  const float bi = bias[0 * H_ + ncol];
  const float bf = bias[1 * H_ + ncol];
  const float bg = bias[2 * H_ + ncol];
  const float bo = bias[3 * H_ + ncol];

  // Per-lane LDS B-fragment base indices (elements)
  int bWhh[4], bWih[4];
#pragma unroll
  for (int g = 0; g < 4; ++g) {
    bWhh[g] = (g * 16 + l15) * WHH_STRIDE + klB;
    bWih[g] = (g * 16 + l15) * WIH_STRIDE + klB;
  }
  const int bWfc = l15 * WFC_STRIDE + klB;

#pragma unroll 1
  for (int t = 0; t < T_; ++t) {
    const unsigned short* hin  = (t & 1) ? hB : hA;
    unsigned short*       hout = (t & 1) ? hA : hB;

    v8f acc[4] = {};

    // gates += x_t @ W_ih^T   (K = I_ = 256; B from LDS)
    const unsigned short* ax = xb + ((size_t)arow * T_ + t) * I_ + klA;
    __builtin_prefetch(ax + I_, 0, 3);   // next step's x row (streaming)
#pragma unroll 2
    for (int k = 0; k < I_; k += 32) {
      v16bf a = loadA(ax + k);
#pragma unroll
      for (int g = 0; g < 4; ++g) {
        v16bf b = loadB16(&wih_sh[bWih[g] + k]);
        acc[g] = __builtin_amdgcn_wmma_f32_16x16x32_bf16(false, a, false, b,
                                                         (short)0, acc[g], false, false);
      }
    }

    // gates += h_t @ W_hh^T   (K = H_ = 1024; B served from LDS, 512-step reuse)
    const unsigned short* ah = hin + (size_t)arow * H_ + klA;
#pragma unroll 2
    for (int k = 0; k < H_; k += 32) {
      v16bf a = loadA(ah + k);
#pragma unroll
      for (int g = 0; g < 4; ++g) {
        v16bf b = loadB16(&whh_sh[bWhh[g] + k]);
        acc[g] = __builtin_amdgcn_wmma_f32_16x16x32_bf16(false, a, false, b,
                                                         (short)0, acc[g], false, false);
      }
    }

    // Elementwise LSTM cell update, fully in-register (acc[g][r] is element
    // (M = r + 8*lhalf, N = l15) of gate tile g per ISA C/D layout).
#pragma unroll
    for (int r = 0; r < 8; ++r) {
      const int brow   = rt * 16 + r + lhalf * 8;
      const size_t idx = (size_t)brow * H_ + ncol;
      float ig = sigm(acc[0][r] + bi);
      float fg = sigm(acc[1][r] + bf);
      float gg = tanhf(acc[2][r] + bg);
      float og = sigm(acc[3][r] + bo);
      float cn = fg * c_state[idx] + ig * gg;
      c_state[idx] = cn;
      hout[idx] = f2bf(og * tanhf(cn));
    }

    // One device-wide barrier per step: h_{t+1} complete before anyone reads it.
    grid_barrier(cnt, gen);

    // Fused per-step FC: tile (frt, fct); K split over 4 waves, reduced via ds_add_f32.
    red[threadIdx.x]       = 0.0f;
    red[threadIdx.x + BLK] = 0.0f;
    __syncthreads();

    v8f facc = {};
    const unsigned short* af = hout + (size_t)(frt * 16 + l15) * H_ + klA;
    const int k0 = wave * 256;
#pragma unroll 2
    for (int k = k0; k < k0 + 256; k += 32) {
      v16bf a = loadA(af + k);
      v16bf b = loadB16(&wfc_sh[bWfc + k]);
      facc = __builtin_amdgcn_wmma_f32_16x16x32_bf16(false, a, false, b,
                                                     (short)0, facc, false, false);
    }
#pragma unroll
    for (int r = 0; r < 8; ++r) {
      atomicAdd(&red[(r + lhalf * 8) * 16 + l15], facc[r]);
    }
    __syncthreads();

    for (int e = threadIdx.x; e < 256; e += BLK) {
      const int m = e >> 4, n = e & 15;
      out[((size_t)t * B_ + (frt * 16 + m)) * I_ + (fct * 16 + n)] = red[e] + bfc[fct * 16 + n];
    }
    // red re-zeroing next iteration is ordered by next grid_barrier's __syncthreads.
  }
}

extern "C" void kernel_launch(void* const* d_in, const int* in_sizes, int n_in,
                              void* d_out, int out_size, void* d_ws, size_t ws_size,
                              hipStream_t stream) {
  (void)in_sizes; (void)n_in; (void)out_size; (void)ws_size;
  const float* x   = (const float*)d_in[0];
  const float* h0  = (const float*)d_in[1];
  const float* c0  = (const float*)d_in[2];
  const float* Wih = (const float*)d_in[3];
  const float* Whh = (const float*)d_in[4];
  const float* bih = (const float*)d_in[5];
  const float* bhh = (const float*)d_in[6];
  const float* Wfc = (const float*)d_in[7];
  const float* bfc = (const float*)d_in[8];
  float* out = (float*)d_out;

  char* w = (char*)d_ws;
  auto carve = [&](size_t bytes) -> char* {
    char* p = w;
    w += (bytes + 255) & ~(size_t)255;
    return p;
  };
  unsigned short* xb   = (unsigned short*)carve((size_t)B_ * T_ * I_ * 2);
  unsigned short* Wihb = (unsigned short*)carve((size_t)G4H * I_ * 2);
  unsigned short* Whhb = (unsigned short*)carve((size_t)G4H * H_ * 2);
  unsigned short* Wfcb = (unsigned short*)carve((size_t)I_ * H_ * 2);
  float*          bias = (float*)carve((size_t)G4H * 4);
  float*          cst  = (float*)carve((size_t)B_ * H_ * 4);
  unsigned short* hAp  = (unsigned short*)carve((size_t)B_ * H_ * 2);
  unsigned short* hBp  = (unsigned short*)carve((size_t)B_ * H_ * 2);
  unsigned*       bar  = (unsigned*)carve(256);

  (void)hipFuncSetAttribute((const void*)lstm_fused,
                            hipFuncAttributeMaxDynamicSharedMemorySize, SMEM_BYTES);

  const int nx = B_ * T_ * I_;
  k_cvt_bf16<<<(nx + 255) / 256, 256, 0, stream>>>(x, xb, nx);
  const int n1 = G4H * I_;
  k_cvt_bf16<<<(n1 + 255) / 256, 256, 0, stream>>>(Wih, Wihb, n1);
  const int n2 = G4H * H_;
  k_cvt_bf16<<<(n2 + 255) / 256, 256, 0, stream>>>(Whh, Whhb, n2);
  const int n3 = I_ * H_;
  k_cvt_bf16<<<(n3 + 255) / 256, 256, 0, stream>>>(Wfc, Wfcb, n3);
  k_prep<<<(B_ * H_ + 255) / 256, 256, 0, stream>>>(bih, bhh, bias, c0, cst, h0, hAp, bar);
  lstm_fused<<<NWG, BLK, SMEM_BYTES, stream>>>(xb, Wihb, Whhb, Wfcb, bias, bfc, cst,
                                               hAp, hBp, out, bar);
}
